// NodeGat_9045201125524
// MI455X (gfx1250) — matrix operand order repaired and checked by reading, not compiled, not run on hardware
//
#include <hip/hip_runtime.h>

typedef __attribute__((ext_vector_type(16))) __bf16 v16bf;
typedef __attribute__((ext_vector_type(8)))  __bf16 v8bf;
typedef __attribute__((ext_vector_type(8)))  float  v8f;

#define NNODES   100000
#define FIN      128
#define HC       256      // heads(4) * hidden(64)
#define NHEAD    4
#define HID      64
#define NCLS     40
#define NEG_SLOPE 0.2f

// ---------- helpers ----------

__device__ __forceinline__ __bf16 f2bf(float f) {
    union { float f; unsigned u; } v; v.f = f;
    unsigned r = (v.u + 0x7FFFu + ((v.u >> 16) & 1u)) >> 16;   // round-nearest-even
    unsigned short s = (unsigned short)r;
    return __builtin_bit_cast(__bf16, s);
}

// float atomic max via signed/unsigned integer ordering trick
__device__ __forceinline__ void atomicMaxF(float* addr, float val) {
    if (val >= 0.0f) atomicMax((int*)addr, __float_as_int(val));
    else             atomicMin((unsigned int*)addr, __float_as_uint(val));
}

__device__ __forceinline__ float leaky(float v) {
    return v > 0.0f ? v : NEG_SLOPE * v;
}

// ---------- generic fills / conversions ----------

__global__ void fill_f32(float* __restrict__ p, float v, long long n) {
    long long i = (long long)blockIdx.x * blockDim.x + threadIdx.x;
    if (i < n) p[i] = v;
}

__global__ void cvt_f32_bf16(const float* __restrict__ in, __bf16* __restrict__ out, long long n) {
    long long i = (long long)blockIdx.x * blockDim.x + threadIdx.x;
    if (i < n) out[i] = f2bf(in[i]);
}

// next-layer input: bf16(relu(agg + bias)), row length HC
__global__ void cvt_bias_relu_bf16(const float* __restrict__ agg, const float* __restrict__ bias,
                                   __bf16* __restrict__ out, long long n) {
    long long i = (long long)blockIdx.x * blockDim.x + threadIdx.x;
    if (i < n) {
        float v = agg[i] + bias[i & (HC - 1)];
        out[i] = f2bf(v > 0.0f ? v : 0.0f);
    }
}

// ---------- pack W [K,256] f32 -> WMMA B-fragment bf16 layout ----------
// Wp[((ks*16 + nt)*32 + lane)*16 + j] ; lane<16: col=nt*16+lane, K=ks*32+(j&7)+16*(j>>3)
//                                      lane>=16: col=nt*16+lane-16, K=ks*32+8+(j&7)+16*(j>>3)
__global__ void pack_w_kernel(const float* __restrict__ W, __bf16* __restrict__ Wp, int ksteps) {
    int tid = blockIdx.x * blockDim.x + threadIdx.x;
    int total = ksteps * 16 * 32;
    if (tid >= total) return;
    int lane = tid & 31;
    int nt   = (tid >> 5) & 15;
    int ks   = tid >> 9;
    int n     = nt * 16 + (lane & 15);
    int kbase = ks * 32 + (lane >> 4) * 8;
    __bf16* dst = Wp + (size_t)tid * 16;
    for (int j = 0; j < 16; ++j)
        dst[j] = f2bf(W[(size_t)(kbase + (j & 7) + ((j >> 3) << 4)) * HC + n]);
}

// ---------- GEMM: H[N,256] f32 = A[N,K] bf16 @ W (packed) via v_wmma_f32_16x16x32_bf16 ----------
// block = 128 threads (4 waves) = one 16-row M tile; wave w owns columns [w*64, w*64+64)
template <int K>
__global__ __launch_bounds__(128) void gemm_kernel(const __bf16* __restrict__ A,
                                                   const __bf16* __restrict__ Wp,
                                                   float* __restrict__ H) {
    const int lane  = threadIdx.x & 31;
    const int wave  = threadIdx.x >> 5;
    const int khalf = lane >> 4;
    const int mrow  = (blockIdx.x << 4) + (lane & 15);
    const __bf16* arow = A + (size_t)mrow * K + khalf * 8;

    v8f acc[4];
    for (int j = 0; j < 4; ++j)
        for (int r = 0; r < 8; ++r) acc[j][r] = 0.0f;

    for (int ks = 0; ks < K / 32; ++ks) {
        // A fragment (16-bit A 16x32 layout): elems 0-7 = K[khalf*8 ..], elems 8-15 = +16
        v8bf alo = *(const v8bf*)(arow + ks * 32);
        v8bf ahi = *(const v8bf*)(arow + ks * 32 + 16);
        v16bf afrag = __builtin_shufflevector(alo, ahi, 0,1,2,3,4,5,6,7,8,9,10,11,12,13,14,15);
        const __bf16* wbase = Wp + ((size_t)((ks * 16 + wave * 4) * 32 + lane)) * 16;
#pragma unroll
        for (int j = 0; j < 4; ++j) {
            v16bf bfrag = *(const v16bf*)(wbase + (size_t)j * 32 * 16);
            acc[j] = __builtin_amdgcn_wmma_f32_16x16x32_bf16(
                false, afrag, false, bfrag, (short)0, acc[j], false, false);
        }
    }

    // D layout: VGPR r -> (M=r, N=lane) for lanes 0-15 ; (M=r+8, N=lane-16) for lanes 16-31
    const int col0  = wave * 64 + (lane & 15);
    const int mbase = (blockIdx.x << 4) + khalf * 8;
    for (int j = 0; j < 4; ++j)
        for (int r = 0; r < 8; ++r)
            H[(size_t)(mbase + r) * HC + col0 + j * 16] = acc[j][r];
}

// ---------- per-node attention coefficients: al[n,h] = sum_c h[n,h*64+c]*a[h,c] ----------
// one wave per node; lane covers channels (lane, lane+32) per head
__global__ void attn_kernel(const float* __restrict__ h,
                            const float* __restrict__ asrc, const float* __restrict__ adst,
                            float* __restrict__ als, float* __restrict__ ald) {
    int lane = threadIdx.x & 31;
    int wave = threadIdx.x >> 5;
    int n = blockIdx.x * 8 + wave;
    if (n >= NNODES) return;
    const float* hr = h + (size_t)n * HC;
    for (int head = 0; head < NHEAD; ++head) {
        int c0 = head * HID + lane;
        float s = hr[c0] * asrc[head * HID + lane] + hr[c0 + 32] * asrc[head * HID + lane + 32];
        float d = hr[c0] * adst[head * HID + lane] + hr[c0 + 32] * adst[head * HID + lane + 32];
        for (int m = 16; m >= 1; m >>= 1) {
            s += __shfl_xor(s, m, 32);
            d += __shfl_xor(d, m, 32);
        }
        if (lane == 0) { als[n * NHEAD + head] = s; ald[n * NHEAD + head] = d; }
    }
}

// ---------- edge kernels (self-loops appended: e>=E -> node e-E) ----------

__global__ void edge_max_kernel(const int* __restrict__ src, const int* __restrict__ dst, int E, int Etot,
                                const float* __restrict__ als, const float* __restrict__ ald,
                                float* __restrict__ m) {
    int e = blockIdx.x * blockDim.x + threadIdx.x;
    if (e >= Etot) return;
    int s = (e < E) ? src[e] : (e - E);
    int d = (e < E) ? dst[e] : (e - E);
    for (int head = 0; head < NHEAD; ++head) {
        float v = leaky(als[s * NHEAD + head] + ald[d * NHEAD + head]);
        atomicMaxF(&m[d * NHEAD + head], v);
    }
}

__global__ void edge_exp_kernel(const int* __restrict__ src, const int* __restrict__ dst, int E, int Etot,
                                const float* __restrict__ als, const float* __restrict__ ald,
                                const float* __restrict__ m, float* __restrict__ p, float* __restrict__ z) {
    int e = blockIdx.x * blockDim.x + threadIdx.x;
    if (e >= Etot) return;
    int s = (e < E) ? src[e] : (e - E);
    int d = (e < E) ? dst[e] : (e - E);
    for (int head = 0; head < NHEAD; ++head) {
        float v = leaky(als[s * NHEAD + head] + ald[d * NHEAD + head]);
        float pe = __expf(v - m[d * NHEAD + head]);
        p[(size_t)e * NHEAD + head] = pe;
        atomicAdd(&z[d * NHEAD + head], pe);
    }
}

// one block (256 thr) per edge: agg[dst, c] += h[src, c] * alpha[head(c)]
__global__ void aggregate_kernel(const int* __restrict__ src, const int* __restrict__ dst, int E,
                                 const float* __restrict__ h, const float* __restrict__ p,
                                 const float* __restrict__ z, float* __restrict__ agg) {
    int e = blockIdx.x;
    int c = threadIdx.x;
    int s = (e < E) ? src[e] : (e - E);
    int d = (e < E) ? dst[e] : (e - E);
    int head = c >> 6;
    float alpha = p[(size_t)e * NHEAD + head] / z[d * NHEAD + head];
    atomicAdd(&agg[(size_t)d * HC + c], h[(size_t)s * HC + c] * alpha);
}

// ---------- classifier + log_softmax: out[n,:] = logsoftmax(relu(agg+b2) @ Wc + bc) ----------
__global__ void classifier_kernel(const float* __restrict__ agg, const float* __restrict__ b2,
                                  const float* __restrict__ Wc, const float* __restrict__ bc,
                                  float* __restrict__ out) {
    __shared__ float inbuf[HC];
    __shared__ float lg[NCLS];
    int n = blockIdx.x;
    int t = threadIdx.x;          // blockDim = 64
    for (int i = 0; i < HC / 64; ++i) {
        int c = t + 64 * i;
        float v = agg[(size_t)n * HC + c] + b2[c];
        inbuf[c] = v > 0.0f ? v : 0.0f;
    }
    __syncthreads();
    if (t < NCLS) {
        float acc = bc[t];
        for (int c = 0; c < HC; ++c) acc += inbuf[c] * Wc[(size_t)c * NCLS + t];
        lg[t] = acc;
    }
    __syncthreads();
    float mx = -3.0e38f;
    for (int j = 0; j < NCLS; ++j) mx = fmaxf(mx, lg[j]);
    float sum = 0.0f;
    for (int j = 0; j < NCLS; ++j) sum += __expf(lg[j] - mx);
    float lse = __logf(sum) + mx;
    if (t < NCLS) out[(size_t)n * NCLS + t] = lg[t] - lse;
}

// ---------- host ----------

static inline size_t align256(size_t x) { return (x + 255) & ~(size_t)255; }

extern "C" void kernel_launch(void* const* d_in, const int* in_sizes, int n_in,
                              void* d_out, int out_size, void* d_ws, size_t ws_size,
                              hipStream_t stream) {
    (void)n_in; (void)out_size; (void)ws_size;
    const float* x   = (const float*)d_in[0];
    const int*   ei  = (const int*)d_in[1];
    const float* W1  = (const float*)d_in[2];
    const float* a1s = (const float*)d_in[3];
    const float* a1d = (const float*)d_in[4];
    const float* b1  = (const float*)d_in[5];
    const float* W2  = (const float*)d_in[6];
    const float* a2s = (const float*)d_in[7];
    const float* a2d = (const float*)d_in[8];
    const float* b2  = (const float*)d_in[9];
    const float* Wc  = (const float*)d_in[10];
    const float* bc  = (const float*)d_in[11];
    float* out = (float*)d_out;

    const int E    = in_sizes[1] / 2;
    const int Etot = E + NNODES;
    const int* src = ei;
    const int* dst = ei + E;

    char* ws = (char*)d_ws;
    size_t off = 0;
    float*  h    = (float*) (ws + off); off = align256(off + (size_t)NNODES * HC * 4);
    float*  agg  = (float*) (ws + off); off = align256(off + (size_t)NNODES * HC * 4);
    __bf16* abf  = (__bf16*)(ws + off); off = align256(off + (size_t)NNODES * HC * 2);
    float*  p    = (float*) (ws + off); off = align256(off + (size_t)Etot * NHEAD * 4);
    float*  als  = (float*) (ws + off); off = align256(off + (size_t)NNODES * NHEAD * 4);
    float*  ald  = (float*) (ws + off); off = align256(off + (size_t)NNODES * NHEAD * 4);
    float*  mbuf = (float*) (ws + off); off = align256(off + (size_t)NNODES * NHEAD * 4);
    float*  zbuf = (float*) (ws + off); off = align256(off + (size_t)NNODES * NHEAD * 4);
    __bf16* wp1  = (__bf16*)(ws + off); off = align256(off + (size_t)(FIN / 32) * 16 * 32 * 16 * 2);
    __bf16* wp2  = (__bf16*)(ws + off); off = align256(off + (size_t)(HC  / 32) * 16 * 32 * 16 * 2);

    const int TPB = 256;
    const long long nvec  = (long long)NNODES * NHEAD;
    const long long nfeat = (long long)NNODES * HC;
    const int egrid = (Etot + TPB - 1) / TPB;
    const int MTILES = NNODES / 16;           // 6250

    // weight packing
    pack_w_kernel<<<(FIN / 32 * 16 * 32 + TPB - 1) / TPB, TPB, 0, stream>>>(W1, wp1, FIN / 32);
    pack_w_kernel<<<(HC  / 32 * 16 * 32 + TPB - 1) / TPB, TPB, 0, stream>>>(W2, wp2, HC / 32);

    // ---- layer 1 ----
    cvt_f32_bf16<<<(int)(((long long)NNODES * FIN + TPB - 1) / TPB), TPB, 0, stream>>>(x, abf, (long long)NNODES * FIN);
    gemm_kernel<FIN><<<MTILES, 128, 0, stream>>>(abf, wp1, h);
    attn_kernel<<<(NNODES + 7) / 8, 256, 0, stream>>>(h, a1s, a1d, als, ald);
    fill_f32<<<(int)((nvec + TPB - 1) / TPB), TPB, 0, stream>>>(mbuf, -3.0e38f, nvec);
    fill_f32<<<(int)((nvec + TPB - 1) / TPB), TPB, 0, stream>>>(zbuf, 0.0f, nvec);
    fill_f32<<<(int)((nfeat + TPB - 1) / TPB), TPB, 0, stream>>>(agg, 0.0f, nfeat);
    edge_max_kernel<<<egrid, TPB, 0, stream>>>(src, dst, E, Etot, als, ald, mbuf);
    edge_exp_kernel<<<egrid, TPB, 0, stream>>>(src, dst, E, Etot, als, ald, mbuf, p, zbuf);
    aggregate_kernel<<<Etot, HC, 0, stream>>>(src, dst, E, h, p, zbuf, agg);

    // ---- layer 2 ----
    cvt_bias_relu_bf16<<<(int)((nfeat + TPB - 1) / TPB), TPB, 0, stream>>>(agg, b1, abf, nfeat);
    gemm_kernel<HC><<<MTILES, 128, 0, stream>>>(abf, wp2, h);
    attn_kernel<<<(NNODES + 7) / 8, 256, 0, stream>>>(h, a2s, a2d, als, ald);
    fill_f32<<<(int)((nvec + TPB - 1) / TPB), TPB, 0, stream>>>(mbuf, -3.0e38f, nvec);
    fill_f32<<<(int)((nvec + TPB - 1) / TPB), TPB, 0, stream>>>(zbuf, 0.0f, nvec);
    fill_f32<<<(int)((nfeat + TPB - 1) / TPB), TPB, 0, stream>>>(agg, 0.0f, nfeat);
    edge_max_kernel<<<egrid, TPB, 0, stream>>>(src, dst, E, Etot, als, ald, mbuf);
    edge_exp_kernel<<<egrid, TPB, 0, stream>>>(src, dst, E, Etot, als, ald, mbuf, p, zbuf);
    aggregate_kernel<<<Etot, HC, 0, stream>>>(src, dst, E, h, p, zbuf, agg);

    // ---- classifier + log_softmax ----
    classifier_kernel<<<NNODES, 64, 0, stream>>>(agg, b2, Wc, bc, out);
}
